// gnn_encoder_20091857011044
// MI455X (gfx1250) — compile-verified, matrix operand order checked by reading, not run on hardware
//
#include <hip/hip_runtime.h>
#include <hip/hip_bf16.h>
#include <stdint.h>

#define NN     50000
#define EE     800000
#define FEDGE  3
#define HID    16
#define H1     128
#define H2     64
#define W3OUT  256
#define NEGS   0.2f
#define PERSIST_WG 1024            // persistent blocks for the edge kernel

typedef __attribute__((ext_vector_type(16))) _Float16 v16h;
typedef __attribute__((ext_vector_type(8)))  float    v8f;

__device__ __forceinline__ v8f wmma16(v16h a, v16h b, v8f c) {
  // v_wmma_f32_16x16x32_f16  (8-arg CDNA5 form)
  return __builtin_amdgcn_wmma_f32_16x16x32_f16(false, a, false, b, (short)0, c, false, false);
}

// Load a 16x32 f16 fragment (A layout; B mirrors with N on lanes) from LDS.
// Row-major tile with given row stride (in halves). kbase = 2v + (v>=4?8:0) + 8*laneHalf.
// kb runs {0,2,4,6,16,18,20,22}+8h -> two contiguous 8-half runs => 2x ds_load_b128.
__device__ __forceinline__ v16h ld_frag(const _Float16* base, int lane, int stride) {
  const int r = lane & 15;
  const int h = lane >> 4;
  const _Float16* p = base + r * stride;
  v16h f;
#pragma unroll
  for (int v = 0; v < 8; ++v) {
    const int kb = 2 * v + ((v & 4) << 1) + (h << 3);
    f[2 * v]     = p[kb];
    f[2 * v + 1] = p[kb + 1];
  }
  return f;
}

__device__ __forceinline__ unsigned fkey(float f) {
  unsigned b = __float_as_uint(f);
  return (b & 0x80000000u) ? ~b : (b | 0x80000000u);
}
__device__ __forceinline__ float fdecode(unsigned k) {
  return __uint_as_float((k & 0x80000000u) ? (k ^ 0x80000000u) : ~k);
}

// ---------------------------------------------------------------------------
// Kernel 0: zero/init workspace accumulators
// ---------------------------------------------------------------------------
__global__ void k_init(float* __restrict__ agg, float* __restrict__ attn,
                       float* __restrict__ z, unsigned* __restrict__ mkey) {
  long long i = (long long)blockIdx.x * blockDim.x + threadIdx.x;
  if (i < (long long)NN * 16) { agg[i] = 0.f; attn[i] = 0.f; }
  if (i < NN) { z[i] = 0.f; mkey[i] = 0u; }
}

// ---------------------------------------------------------------------------
// Kernel 1: fused EdgeMLP (WMMA) + per-edge message + scatter-add
//   persistent blocks: stage f16 weights in LDS once, grid-stride over
//   128-edge tiles (one wave = 16 edges; 8 waves/block; E = 6250*128 exact)
// ---------------------------------------------------------------------------
__global__ __launch_bounds__(256) void k_edge_mlp_msg(
    const float* __restrict__ x, const long long* __restrict__ ei,
    const float* __restrict__ ea,
    const float* __restrict__ w1, const float* __restrict__ b1,
    const float* __restrict__ w2, const float* __restrict__ b2,
    const float* __restrict__ w3, const float* __restrict__ b3,
    float* __restrict__ agg) {
  __shared__ _Float16 w1t[H1 * 32];      // Bt: [n][k], K padded 3->32
  __shared__ _Float16 w2t[H2 * H1];      // Bt: [n][k], K = 128
  __shared__ _Float16 w3t[W3OUT * H2];   // Bt: [n][k], K = 64
  __shared__ float b1s[H1];
  __shared__ float b2s[H2];
  __shared__ float b3s[W3OUT];
  __shared__ _Float16 h1t[8][16 * H1];   // per-wave layer1 activations
  __shared__ _Float16 hxt[8][16 * H2];   // per-wave: edge_attr tile, then layer2
  __shared__ alignas(16) float xst[8][16 * 16]; // gathered x[src]
  __shared__ int idst[8][16];            // dst ids

  const int tid = threadIdx.x;
  // ---- stage weights once per (persistent) block ----
  for (int i = tid; i < H1 * 32; i += 256) {
    int n = i >> 5, k = i & 31;
    w1t[i] = (k < FEDGE) ? (_Float16)w1[k * H1 + n] : (_Float16)0.f;
  }
  for (int i = tid; i < H2 * H1; i += 256) {
    int n = i >> 7, k = i & 127;
    w2t[i] = (_Float16)w2[k * H2 + n];
  }
  for (int i = tid; i < W3OUT * H2; i += 256) {
    int n = i >> 6, k = i & 63;
    w3t[i] = (_Float16)w3[k * W3OUT + n];
  }
  for (int i = tid; i < H1; i += 256) b1s[i] = b1[i];
  for (int i = tid; i < H2; i += 256) b2s[i] = b2[i];
  for (int i = tid; i < W3OUT; i += 256) b3s[i] = b3[i];
  __syncthreads();

  const int wave = tid >> 5;
  const int lane = tid & 31;
  const int n    = lane & 15;   // C-tile column
  const int hh   = lane >> 4;   // C-tile row-half
  const int numBlockTiles = EE / 128;

  for (int bt = blockIdx.x; bt < numBlockTiles; bt += gridDim.x) {
    const long long eBase = ((long long)bt * 8 + wave) * 16;

    // prefetch next tile's edge stream into cache (global_prefetch_b8)
    {
      const long long btn = (long long)(bt + gridDim.x);
      if (btn < numBlockTiles && lane < 16) {
        const long long en = (btn * 8 + wave) * 16 + lane;
        __builtin_prefetch(ea + en * FEDGE, 0, 1);
        __builtin_prefetch(ei + en, 0, 1);
        __builtin_prefetch(ei + EE + en, 0, 1);
      }
    }

    // gather: edge_attr (f16, stride H2, zero-padded K), x[src], dst ids
    if (lane < 16) {
      const long long e = eBase + lane;
      const int s = (int)ei[e];
      idst[wave][lane] = (int)ei[EE + e];
      const float* ar = ea + e * FEDGE;
#pragma unroll
      for (int k = 0; k < 32; ++k)
        hxt[wave][lane * H2 + k] = (k < FEDGE) ? (_Float16)ar[k] : (_Float16)0.f;
      const float4* xr = (const float4*)(x + (long long)s * 16);
      float4* xd = (float4*)&xst[wave][lane * 16];
      xd[0] = xr[0]; xd[1] = xr[1]; xd[2] = xr[2]; xd[3] = xr[3];
    }
    __syncthreads();

    // ---- layer 1: [16x3(->32)] @ [32x128] ----
    {
      const v16h a = ld_frag(hxt[wave], lane, H2);
#pragma unroll
      for (int nt = 0; nt < 8; ++nt) {
        v8f c = {};
        c = wmma16(a, ld_frag(w1t + nt * 16 * 32, lane, 32), c);
        const float bias = b1s[nt * 16 + n];
#pragma unroll
        for (int j = 0; j < 8; ++j) {
          float v = c[j] + bias;
          v = v > 0.f ? v : 0.f;
          h1t[wave][(j + 8 * hh) * H1 + nt * 16 + n] = (_Float16)v;
        }
      }
    }
    __syncthreads();

    // ---- layer 2: [16x128] @ [128x64] ----
    {
      v16h a2[4];
#pragma unroll
      for (int kt = 0; kt < 4; ++kt) a2[kt] = ld_frag(h1t[wave] + kt * 32, lane, H1);
#pragma unroll
      for (int nt = 0; nt < 4; ++nt) {
        v8f c = {};
#pragma unroll
        for (int kt = 0; kt < 4; ++kt)
          c = wmma16(a2[kt], ld_frag(w2t + nt * 16 * H1 + kt * 32, lane, H1), c);
        const float bias = b2s[nt * 16 + n];
#pragma unroll
        for (int j = 0; j < 8; ++j) {
          float v = c[j] + bias;
          v = v > 0.f ? v : 0.f;
          hxt[wave][(j + 8 * hh) * H2 + nt * 16 + n] = (_Float16)v;
        }
      }
    }
    __syncthreads();

    // ---- layer 3 tile-by-tile, fold x[src] into message (never store w_e) ----
    {
      v16h a3[2];
#pragma unroll
      for (int kt = 0; kt < 2; ++kt) a3[kt] = ld_frag(hxt[wave] + kt * 32, lane, H2);
      v8f msg = {};
#pragma unroll
      for (int t = 0; t < 16; ++t) {        // output cols 16t..16t+15 == w_e[:, i=t, :]
        v8f c = {};
#pragma unroll
        for (int kt = 0; kt < 2; ++kt)
          c = wmma16(a3[kt], ld_frag(w3t + t * 16 * H2 + kt * 32, lane, H2), c);
        const float bias = b3s[t * 16 + n];
#pragma unroll
        for (int j = 0; j < 8; ++j) {
          const int row = j + 8 * hh;
          msg[j] += xst[wave][row * 16 + t] * (c[j] + bias);
        }
      }
#pragma unroll
      for (int j = 0; j < 8; ++j) {
        const int row = j + 8 * hh;
        const int d = idst[wave][row];
        atomicAdd(&agg[(long long)d * HID + n], msg[j]);
      }
    }
    __syncthreads();   // protect per-wave tiles before next iteration's gather
  }
}

// ---------------------------------------------------------------------------
// Kernel 2: x1 = relu(agg + x@root_w + nn_bias); xt = x1 @ gat_w  (WMMA)
// ---------------------------------------------------------------------------
__global__ __launch_bounds__(256) void k_node_x1_xt(
    const float* __restrict__ x, const float* __restrict__ agg,
    const float* __restrict__ root_w, const float* __restrict__ nn_bias,
    const float* __restrict__ gat_w, float* __restrict__ xt) {
  __shared__ _Float16 rwt[16 * 32];
  __shared__ _Float16 gwt[16 * 32];
  __shared__ _Float16 tile[8][16 * 32];
  const int tid = threadIdx.x;
  for (int i = tid; i < 16 * 32; i += 256) {
    int nn = i >> 5, k = i & 31;
    rwt[i] = (k < 16) ? (_Float16)root_w[k * 16 + nn] : (_Float16)0.f;
    gwt[i] = (k < 16) ? (_Float16)gat_w[k * 16 + nn] : (_Float16)0.f;
  }
  __syncthreads();
  const int wave = tid >> 5, lane = tid & 31;
  const int n = lane & 15, hh = lane >> 4;
  int t = blockIdx.x * 8 + wave;
  if (t >= NN / 16) t = 0;           // duplicate tile 0: identical writes, uniform barriers
  const int base = t * 16;
  if (lane < 16) {
#pragma unroll
    for (int k = 0; k < 32; ++k)
      tile[wave][lane * 32 + k] =
          (k < 16) ? (_Float16)x[(long long)(base + lane) * 16 + k] : (_Float16)0.f;
  }
  __syncthreads();
  v8f c1 = {};
  c1 = wmma16(ld_frag(tile[wave], lane, 32), ld_frag(rwt, lane, 32), c1);
  float x1v[8];
#pragma unroll
  for (int j = 0; j < 8; ++j) {
    const int node = base + j + 8 * hh;
    float v = c1[j] + agg[(long long)node * 16 + n] + nn_bias[n];
    x1v[j] = v > 0.f ? v : 0.f;
  }
  __syncthreads();
#pragma unroll
  for (int j = 0; j < 8; ++j)
    tile[wave][(j + 8 * hh) * 32 + n] = (_Float16)x1v[j];   // pad cols stay 0
  __syncthreads();
  v8f c2 = {};
  c2 = wmma16(ld_frag(tile[wave], lane, 32), ld_frag(gwt, lane, 32), c2);
#pragma unroll
  for (int j = 0; j < 8; ++j)
    xt[(long long)(base + j + 8 * hh) * 16 + n] = c2[j];
}

// ---------------------------------------------------------------------------
// Kernel 3: a_src/a_dst = xt . att
// ---------------------------------------------------------------------------
__global__ void k_attn_coef(const float* __restrict__ xt,
                            const float* __restrict__ att_src,
                            const float* __restrict__ att_dst,
                            float* __restrict__ asrc, float* __restrict__ adst) {
  int i = blockIdx.x * blockDim.x + threadIdx.x;
  if (i >= NN) return;
  const float* r = xt + (long long)i * 16;
  float s = 0.f, d = 0.f;
#pragma unroll
  for (int k = 0; k < 16; ++k) { s += r[k] * att_src[k]; d += r[k] * att_dst[k]; }
  asrc[i] = s; adst[i] = d;
}

// ---------------------------------------------------------------------------
// Kernel 4: segment-max via order-preserving uint atomicMax (edges + self loops)
// ---------------------------------------------------------------------------
__global__ void k_gat_max(const long long* __restrict__ ei,
                          const float* __restrict__ asrc, const float* __restrict__ adst,
                          unsigned* __restrict__ mkey, float* __restrict__ eval) {
  long long i = (long long)blockIdx.x * blockDim.x + threadIdx.x;
  if (i >= (long long)EE + NN) return;
  int s, d;
  if (i < EE) { s = (int)ei[i]; d = (int)ei[EE + i]; }
  else        { s = (int)(i - EE); d = s; }
  float v = asrc[s] + adst[d];
  v = v > 0.f ? v : NEGS * v;
  eval[i] = v;
  atomicMax(&mkey[d], fkey(v));
}

// ---------------------------------------------------------------------------
// Kernel 5: unnormalized softmax scatter: z += w;  attn[d] += w * xt[s]
//   (normalization deferred to kernel 6: out/z == segment_sum(alpha*xt))
// ---------------------------------------------------------------------------
__global__ void k_gat_acc(const long long* __restrict__ ei,
                          const float* __restrict__ eval, const unsigned* __restrict__ mkey,
                          const float* __restrict__ xt,
                          float* __restrict__ z, float* __restrict__ attn) {
  long long i = (long long)blockIdx.x * blockDim.x + threadIdx.x;
  if (i >= (long long)(EE + NN) * 16) return;
  const long long e = i >> 4;
  const int c = (int)(i & 15);
  int s, d;
  if (e < EE) { s = (int)ei[e]; d = (int)ei[EE + e]; }
  else        { s = (int)(e - EE); d = s; }
  const float w = __expf(eval[e] - fdecode(mkey[d]));
  if (c == 0) atomicAdd(&z[d], w);
  atomicAdd(&attn[(long long)d * 16 + c], w * xt[(long long)s * 16 + c]);
}

// ---------------------------------------------------------------------------
// Kernel 6: x2 = relu(attn/z + gat_bias); out = relu(x2 @ fc1_w + fc1_b)  (WMMA)
// ---------------------------------------------------------------------------
__global__ __launch_bounds__(256) void k_out(
    const float* __restrict__ attn, const float* __restrict__ z,
    const float* __restrict__ gat_bias,
    const float* __restrict__ fc1_w, const float* __restrict__ fc1_b,
    float* __restrict__ out) {
  __shared__ _Float16 fwt[64 * 32];
  __shared__ _Float16 tile[8][16 * 32];
  const int tid = threadIdx.x;
  for (int i = tid; i < 64 * 32; i += 256) {
    int nn = i >> 5, k = i & 31;
    fwt[i] = (k < 16) ? (_Float16)fc1_w[k * 64 + nn] : (_Float16)0.f;
  }
  __syncthreads();
  const int wave = tid >> 5, lane = tid & 31;
  const int n = lane & 15, hh = lane >> 4;
  int t = blockIdx.x * 8 + wave;
  if (t >= NN / 16) t = 0;
  const int base = t * 16;
  if (lane < 16) {
    const int node = base + lane;
    const float inv = 1.f / z[node];   // self-loop guarantees z > 0
#pragma unroll
    for (int k = 0; k < 32; ++k) {
      float v = 0.f;
      if (k < 16) {
        v = attn[(long long)node * 16 + k] * inv + gat_bias[k];
        v = v > 0.f ? v : 0.f;
      }
      tile[wave][lane * 32 + k] = (_Float16)v;
    }
  }
  __syncthreads();
  const v16h a = ld_frag(tile[wave], lane, 32);
#pragma unroll
  for (int nt = 0; nt < 4; ++nt) {
    v8f c = {};
    c = wmma16(a, ld_frag(fwt + nt * 16 * 32, lane, 32), c);
#pragma unroll
    for (int j = 0; j < 8; ++j) {
      const int node = base + j + 8 * hh;
      float v = c[j] + fc1_b[nt * 16 + n];
      v = v > 0.f ? v : 0.f;
      out[(long long)node * 64 + nt * 16 + n] = v;
    }
  }
}

// ---------------------------------------------------------------------------
extern "C" void kernel_launch(void* const* d_in, const int* in_sizes, int n_in,
                              void* d_out, int out_size, void* d_ws, size_t ws_size,
                              hipStream_t stream) {
  (void)in_sizes; (void)n_in; (void)out_size; (void)ws_size;
  const float*     x        = (const float*)d_in[0];
  const long long* ei       = (const long long*)d_in[1];  // int64 [2,E]
  const float*     ea       = (const float*)d_in[2];
  // d_in[3] = batch (unused)
  const float* w1       = (const float*)d_in[4];
  const float* b1       = (const float*)d_in[5];
  const float* w2       = (const float*)d_in[6];
  const float* b2       = (const float*)d_in[7];
  const float* w3       = (const float*)d_in[8];
  const float* b3       = (const float*)d_in[9];
  const float* root_w   = (const float*)d_in[10];
  const float* nn_bias  = (const float*)d_in[11];
  const float* gat_w    = (const float*)d_in[12];
  const float* att_src  = (const float*)d_in[13];
  const float* att_dst  = (const float*)d_in[14];
  const float* gat_bias = (const float*)d_in[15];
  const float* fc1_w    = (const float*)d_in[16];
  const float* fc1_b    = (const float*)d_in[17];

  float* ws   = (float*)d_ws;
  float* agg  = ws;                           // N*16
  float* xt   = agg  + (size_t)NN * 16;       // N*16
  float* attn = xt   + (size_t)NN * 16;       // N*16
  float* asrc = attn + (size_t)NN * 16;       // N
  float* adst = asrc + NN;                    // N
  float* z    = adst + NN;                    // N
  unsigned* mkey = (unsigned*)(z + NN);       // N
  float* eval = (float*)(mkey + NN);          // E+N
  float* outp = (float*)d_out;

  k_init<<<(NN * 16 + 255) / 256, 256, 0, stream>>>(agg, attn, z, mkey);
  k_edge_mlp_msg<<<PERSIST_WG, 256, 0, stream>>>(x, ei, ea, w1, b1, w2, b2, w3, b3, agg);
  k_node_x1_xt<<<(NN / 16 + 7) / 8, 256, 0, stream>>>(x, agg, root_w, nn_bias, gat_w, xt);
  k_attn_coef<<<(NN + 255) / 256, 256, 0, stream>>>(xt, att_src, att_dst, asrc, adst);
  k_gat_max<<<(EE + NN + 255) / 256, 256, 0, stream>>>(ei, asrc, adst, mkey, eval);
  k_gat_acc<<<(int)(((long long)(EE + NN) * 16 + 255) / 256), 256, 0, stream>>>(ei, eval, mkey, xt, z, attn);
  k_out<<<(NN / 16 + 7) / 8, 256, 0, stream>>>(attn, z, gat_bias, fc1_w, fc1_b, outp);
}
